// GST_LSTM_68436008895010
// MI455X (gfx1250) — compile-verified, matrix operand order checked by reading, not run on hardware
//
#include <hip/hip_runtime.h>
#include <math.h>

#define NROW 16384
#define KDIM 16384
#define HID  64
#define INDIM 64

typedef __attribute__((ext_vector_type(16))) __bf16 v16bf;
typedef __attribute__((ext_vector_type(8)))  __bf16 bf16x8;
typedef __attribute__((ext_vector_type(4)))  __bf16 bf16x4;
typedef __attribute__((ext_vector_type(4)))  float  f32x4;
typedef __attribute__((ext_vector_type(8)))  float  v8f;

// Fragment overlay: v16bf per-lane layout is two contiguous 8-element runs:
//   a[0..7]  = A[m][half*8 + 0..7]
//   a[8..15] = A[m][16 + half*8 + 0..7]
union Frag {
  v16bf  v;
  bf16x4 q[4];
  bf16x8 o[2];
};

__device__ __forceinline__ bf16x4 cvt4(const float* p) {
  return __builtin_convertvector(*(const f32x4*)p, bf16x4);
}

// ---------------------------------------------------------------------------
// Kernel 1: Ht_1 (N x 64, f32) -> Ht1T (64 x N, bf16)
// ---------------------------------------------------------------------------
__global__ void __launch_bounds__(256) k_transpose_bf16(const float* __restrict__ src,
                                                        __bf16* __restrict__ dst) {
  size_t idx = (size_t)blockIdx.x * 256 + threadIdx.x;   // over N*64
  int i = (int)(idx >> 6);
  int j = (int)(idx & 63);
  dst[(size_t)j * KDIM + i] = (__bf16)src[idx];
}

// ---------------------------------------------------------------------------
// Kernel 2: convert 8 weight matrices (64x64 f32, row-major) -> bf16 packed
// order: Wxi, Whi, Wxf, Whf, Wxo, Who, Wxc, Whc
// ---------------------------------------------------------------------------
__global__ void __launch_bounds__(256) k_convert_weights(
    const float* __restrict__ w0, const float* __restrict__ w1,
    const float* __restrict__ w2, const float* __restrict__ w3,
    const float* __restrict__ w4, const float* __restrict__ w5,
    const float* __restrict__ w6, const float* __restrict__ w7,
    __bf16* __restrict__ out) {
  const float* ws[8] = {w0, w1, w2, w3, w4, w5, w6, w7};
  const float* src = ws[blockIdx.x];
  __bf16* dst = out + (size_t)blockIdx.x * 4096;
  for (int i = threadIdx.x * 4; i < 4096; i += 256 * 4)
    *(bf16x4*)(dst + i) = cvt4(src + i);
}

// ---------------------------------------------------------------------------
// Kernel 3: g = adj @ Ht_1 via bf16 WMMA, f32 accumulate, bf16 output.
// Block = 128 thr = 4 waves; each wave owns a 16-row strip and ALL 4 column
// tiles (one adj conversion feeds 4 WMMAs). adj streamed from HBM exactly once.
// ---------------------------------------------------------------------------
__global__ void __launch_bounds__(128) k_gemm_adj(const float* __restrict__ adj,
                                                  const __bf16* __restrict__ ht1t,
                                                  __bf16* __restrict__ g) {
  const int tid  = threadIdx.x;
  const int wave = tid >> 5;
  const int lane = tid & 31;
  const int half = lane >> 4;
  const int ml   = lane & 15;

  const int rowBase = blockIdx.x * 64 + wave * 16;
  const float*  arow = adj  + (size_t)(rowBase + ml) * KDIM + half * 8;
  const __bf16* brow = ht1t + (size_t)ml * KDIM + half * 8;   // + t*16*KDIM per tile

  v8f acc[4] = {{}, {}, {}, {}};

  for (int kk = 0; kk < KDIM; kk += 32) {
    Frag A;
    A.q[0] = cvt4(arow + kk);
    A.q[1] = cvt4(arow + kk + 4);
    A.q[2] = cvt4(arow + kk + 16);
    A.q[3] = cvt4(arow + kk + 20);
    __builtin_prefetch(arow + kk + 512, 0, 1);   // stream adj ahead

#pragma unroll
    for (int t = 0; t < 4; ++t) {
      Frag B;
      const __bf16* bp = brow + (size_t)t * 16 * KDIM + kk;
      B.o[0] = *(const bf16x8*)(bp);
      B.o[1] = *(const bf16x8*)(bp + 16);
      acc[t] = __builtin_amdgcn_wmma_f32_16x16x32_bf16(
          false, A.v, false, B.v, (short)0, acc[t], false, false);
    }
  }

#pragma unroll
  for (int t = 0; t < 4; ++t) {
    __bf16* gout = g + (size_t)rowBase * HID + t * 16 + ml;
#pragma unroll
    for (int r = 0; r < 8; ++r)
      gout[(size_t)(r + 8 * half) * HID] = (__bf16)acc[t][r];
  }
}

// ---------------------------------------------------------------------------
// Kernel 4: fused gates. Block = 16 rows, 8 waves; each wave does 2
// (gate, colTile) tiles; acc = x@Wx^T + g@Wh^T (4 WMMAs/tile). Pre-acts go to
// LDS, then elementwise sigmoid/tanh/Ct/Ht.
// ---------------------------------------------------------------------------
__global__ void __launch_bounds__(256) k_gates(
    const float* __restrict__ ht, const __bf16* __restrict__ gmat,
    const __bf16* __restrict__ wts, const float* __restrict__ Ct_1,
    const float* __restrict__ bxi, const float* __restrict__ bhi,
    const float* __restrict__ bxf, const float* __restrict__ bhf,
    const float* __restrict__ bxo, const float* __restrict__ bho,
    const float* __restrict__ bxc, const float* __restrict__ bhc,
    float* __restrict__ out) {
  __shared__ float pre[4][16][64];   // [gate][m][n], 16 KB

  const int tid  = threadIdx.x;
  const int wave = tid >> 5;
  const int lane = tid & 31;
  const int half = lane >> 4;
  const int ml   = lane & 15;
  const int rowBase = blockIdx.x * 16;

  const float*  xrow = ht   + (size_t)(rowBase + ml) * INDIM + half * 8;
  const __bf16* grow = gmat + (size_t)(rowBase + ml) * HID   + half * 8;

  // A-fragments shared by both tiles this wave computes (K = 64 -> 2 chunks)
  Frag ax[2], ag[2];
#pragma unroll
  for (int kk = 0; kk < 2; ++kk) {
    ax[kk].q[0] = cvt4(xrow + kk * 32);
    ax[kk].q[1] = cvt4(xrow + kk * 32 + 4);
    ax[kk].q[2] = cvt4(xrow + kk * 32 + 16);
    ax[kk].q[3] = cvt4(xrow + kk * 32 + 20);
    ag[kk].o[0] = *(const bf16x8*)(grow + kk * 32);
    ag[kk].o[1] = *(const bf16x8*)(grow + kk * 32 + 16);
  }

#pragma unroll
  for (int t = 0; t < 2; ++t) {
    const int tile = wave * 2 + t;
    const int gate = tile >> 2;   // 0=i,1=f,2=o,3=c
    const int col  = tile & 3;
    const __bf16* wx = wts + (size_t)(2 * gate)     * 4096
                     + (size_t)(col * 16 + ml) * 64 + half * 8;
    const __bf16* wh = wts + (size_t)(2 * gate + 1) * 4096
                     + (size_t)(col * 16 + ml) * 64 + half * 8;

    v8f acc = {};
#pragma unroll
    for (int kk = 0; kk < 2; ++kk) {
      Frag B;
      B.o[0] = *(const bf16x8*)(wx + kk * 32);
      B.o[1] = *(const bf16x8*)(wx + kk * 32 + 16);
      acc = __builtin_amdgcn_wmma_f32_16x16x32_bf16(
          false, ax[kk].v, false, B.v, (short)0, acc, false, false);
    }
#pragma unroll
    for (int kk = 0; kk < 2; ++kk) {
      Frag B;
      B.o[0] = *(const bf16x8*)(wh + kk * 32);
      B.o[1] = *(const bf16x8*)(wh + kk * 32 + 16);
      acc = __builtin_amdgcn_wmma_f32_16x16x32_bf16(
          false, ag[kk].v, false, B.v, (short)0, acc, false, false);
    }

#pragma unroll
    for (int r = 0; r < 8; ++r)
      pre[gate][r + 8 * half][col * 16 + ml] = acc[r];
  }

  __syncthreads();

#pragma unroll
  for (int e = tid; e < 1024; e += 256) {
    const int m = e >> 6, n = e & 63;
    const size_t idx = (size_t)(rowBase + m) * HID + n;
    const float ip = pre[0][m][n] + bxi[n] + bhi[n];
    const float fp = pre[1][m][n] + bxf[n] + bhf[n];
    const float op = pre[2][m][n] + bxo[n] + bho[n];
    const float up = pre[3][m][n] + bxc[n] + bhc[n];
    const float it = 1.f / (1.f + expf(-ip));
    const float ft = 1.f / (1.f + expf(-fp));
    const float ot = 1.f / (1.f + expf(-op));
    const float ut = tanhf(up);
    const float Ct = ft * Ct_1[idx] + it * ut;
    const float Ht = ot * tanhf(Ct);
    out[idx] = Ht;
    out[(size_t)NROW * HID + idx] = Ct;
  }
}

// ---------------------------------------------------------------------------
extern "C" void kernel_launch(void* const* d_in, const int* in_sizes, int n_in,
                              void* d_out, int out_size, void* d_ws, size_t ws_size,
                              hipStream_t stream) {
  const float* ht    = (const float*)d_in[0];
  const float* Ht_1  = (const float*)d_in[1];
  const float* Ct_1  = (const float*)d_in[2];
  const float* adj   = (const float*)d_in[3];

  // workspace layout: [0,2MB) Ht1T bf16; [2MB,4MB) g bf16; [4MB,+64KB) weights bf16
  char* ws = (char*)d_ws;
  __bf16* ht1t = (__bf16*)ws;
  __bf16* g    = (__bf16*)(ws + ((size_t)2 << 20));
  __bf16* wts  = (__bf16*)(ws + ((size_t)4 << 20));

  k_transpose_bf16<<<(NROW * HID) / 256, 256, 0, stream>>>(Ht_1, ht1t);

  k_convert_weights<<<8, 256, 0, stream>>>(
      (const float*)d_in[4],  (const float*)d_in[6],    // Wxi, Whi
      (const float*)d_in[8],  (const float*)d_in[10],   // Wxf, Whf
      (const float*)d_in[12], (const float*)d_in[14],   // Wxo, Who
      (const float*)d_in[16], (const float*)d_in[18],   // Wxc, Whc
      wts);

  k_gemm_adj<<<NROW / 64, 128, 0, stream>>>(adj, ht1t, g);

  k_gates<<<NROW / 16, 256, 0, stream>>>(
      ht, g, wts, Ct_1,
      (const float*)d_in[5],  (const float*)d_in[7],
      (const float*)d_in[9],  (const float*)d_in[11],
      (const float*)d_in[13], (const float*)d_in[15],
      (const float*)d_in[17], (const float*)d_in[19],
      (float*)d_out);
}
